// HierarchicalMoEHiR_63178968924523
// MI455X (gfx1250) — matrix operand order repaired and checked by reading, not compiled, not run on hardware
//
#include <hip/hip_runtime.h>

// ---------------------------------------------------------------------------
// Types matching the CDNA5 WMMA builtins
// ---------------------------------------------------------------------------
typedef __attribute__((ext_vector_type(16))) __bf16 v16bf;
typedef __attribute__((ext_vector_type(8)))  float  v8f;
typedef __attribute__((ext_vector_type(4)))  unsigned int u32x4;
typedef __attribute__((ext_vector_type(4)))  int          i32x4;
typedef __attribute__((ext_vector_type(8)))  int          i32x8;

// Problem constants
#define NT      16384      // B*T tokens
#define DDIM    256
#define FEMB    64
#define NB      4
#define ES      4
#define NE      16
#define DEH     128
#define DRH     64
#define FTOT    32

__device__ __forceinline__ float gelu_erf(float x) {
    return 0.5f * x * (1.0f + erff(x * 0.70710678118654752440f));
}

__device__ __forceinline__ v8f wmma_bf16(v16bf a, v16bf b, v8f c) {
    return __builtin_amdgcn_wmma_f32_16x16x32_bf16(false, a, false, b,
                                                   (short)0, c, false, false);
}

// A-fragment (16x32, row-major bf16 source, stride ldr elements).
// Lane layout (ISA 7.12.2): m = lane%16; ks = half*8+[0,8) and 16+half*8+[0,8).
__device__ __forceinline__ v16bf loadA_bf16(const __bf16* p, int ldr, int lane, int kb) {
    const __bf16* r = p + (lane & 15) * ldr + kb * 32 + ((lane >> 4) << 3);
    union { v16bf v; uint4 u[2]; } U;
    U.u[0] = *(const uint4*)(r);
    U.u[1] = *(const uint4*)(r + 16);
    return U.v;
}

// A-fragment from an f32 row-major source, k < Klim valid (zero padded), kb==0.
__device__ __forceinline__ v16bf loadA_f32pad(const float* rowbase, int ldr, int lane, int Klim) {
    const float* r = rowbase + (lane & 15) * ldr;
    int khalf = (lane >> 4) * 8;
    v16bf o;
#pragma unroll
    for (int j = 0; j < 16; ++j) {
        int k = khalf + ((j < 8) ? j : (8 + j));
        float v = (k < Klim) ? r[k] : 0.0f;
        o[j] = (__bf16)v;
    }
    return o;
}

// B-fragment from a pre-packed 32x16 block (512 bf16): lane's 16 values contiguous.
__device__ __forceinline__ v16bf loadB(const __bf16* blk, int lane) {
    const uint4* q = (const uint4*)(blk + lane * 16);
    union { v16bf v; uint4 u[2]; } U;
    U.u[0] = q[0];
    U.u[1] = q[1];
    return U.v;
}

// ---------------------------------------------------------------------------
// Tensor Data Mover: 1-D copy of 32768 bf16 (64KB) global -> LDS.
// Descriptor per CDNA5 ISA (D# group0/group1 bitfields).
// ---------------------------------------------------------------------------
__device__ __forceinline__ void tdm_copy_64k(const void* gsrc, void* ldst) {
    unsigned long long ga = (unsigned long long)(size_t)gsrc;
    u32x4 g0 = {};
    g0.x = 1u;                                         // count=1, user descriptor
    g0.y = (unsigned int)(size_t)ldst;                 // LDS byte address (low 32 of flat)
    g0.z = (unsigned int)(ga & 0xffffffffull);         // global_addr[31:0]
    g0.w = (unsigned int)((ga >> 32) & 0x1ffffffull)   // global_addr[56:32]
         | (2u << 30);                                 // type=2 ("image")
    i32x8 g1 = {};
    g1[0] = 1 << 16;                                   // data_size=1 (2-byte elements)
    g1[1] = (int)0x80000000u;                          // tensor_dim0 = 32768 (bits[79:48] lo16)
    g1[3] = (int)0x80000000u;                          // tile_dim0   = 32768 (bits[127:112])
    g1[5] = 32768;                                     // tensor_dim0_stride[31:0]
    i32x4 z4 = {};
#if __has_include(<hip/amd_detail/amd_gfx1250_TDM.h>)
    i32x8 z8 = {};
    __builtin_amdgcn_tensor_load_to_lds(g0, g1, z4, z4, z8, 0);
#else
    __builtin_amdgcn_tensor_load_to_lds(g0, g1, z4, z4, 0);
#endif
}

// ---------------------------------------------------------------------------
// Kernel 1: LayerNorm -> bf16 h_norm.  One wave per token (wave32).
// ---------------------------------------------------------------------------
__global__ void __launch_bounds__(256) k_ln(const float* __restrict__ hidden,
                                            const float* __restrict__ g,
                                            const float* __restrict__ b,
                                            __bf16* __restrict__ hbf) {
    int wave = threadIdx.x >> 5, lane = threadIdx.x & 31;
    int t = blockIdx.x * 8 + wave;
    const float* x = hidden + (size_t)t * DDIM + lane * 8;
    float v[8];
#pragma unroll
    for (int i = 0; i < 8; ++i) v[i] = x[i];
    float s = 0.f, s2 = 0.f;
#pragma unroll
    for (int i = 0; i < 8; ++i) { s += v[i]; s2 += v[i] * v[i]; }
#pragma unroll
    for (int m = 16; m >= 1; m >>= 1) {
        s  += __shfl_xor(s,  m, 32);
        s2 += __shfl_xor(s2, m, 32);
    }
    float mean = s * (1.0f / DDIM);
    float var  = s2 * (1.0f / DDIM) - mean * mean;
    float inv  = rsqrtf(var + 1e-5f);
    union { __bf16 h[8]; uint4 q; } U;
#pragma unroll
    for (int i = 0; i < 8; ++i) {
        int d = lane * 8 + i;
        U.h[i] = (__bf16)((v[i] - mean) * inv * g[d] + b[d]);
    }
    *(uint4*)(hbf + (size_t)t * DDIM + lane * 8) = U.q;
}

// ---------------------------------------------------------------------------
// Kernel 2: pack f32 weight [batch][K][N] into WMMA B-fragment blocks:
// block (b,nt,kb) = 512 bf16; element [lane*16+j] = W[kb*32+(lane/16)*8+(j<8?j:j+8)][nt*16+lane%16]
// (zero padded past K,N).
// ---------------------------------------------------------------------------
__global__ void k_pack(const float* __restrict__ src, __bf16* __restrict__ dst,
                       int K, int N, int nkb, int nnt, int batch) {
    int total = batch * nnt * nkb * 512;
    for (int idx = blockIdx.x * blockDim.x + threadIdx.x; idx < total;
         idx += gridDim.x * blockDim.x) {
        int j    = idx & 15;
        int lane = (idx >> 4) & 31;
        int blk  = idx >> 9;
        int kb   = blk % nkb;
        int nt   = (blk / nkb) % nnt;
        int b    = blk / (nkb * nnt);
        int k = kb * 32 + (lane >> 4) * 8 + ((j < 8) ? j : (j + 8));
        int n = nt * 16 + (lane & 15);
        float v = (k < K && n < N) ? src[((size_t)b * K + k) * N + n] : 0.0f;
        dst[idx] = (__bf16)v;
    }
}

// top-2-of-4 softmax, matching jnp where(s >= thr) semantics (ties kept)
__device__ __forceinline__ void top2sm4(const float* s, float* w) {
    int im = 0;
    float mx = s[0];
#pragma unroll
    for (int j = 1; j < 4; ++j) if (s[j] > mx) { mx = s[j]; im = j; }
    float thr = -INFINITY;
#pragma unroll
    for (int j = 0; j < 4; ++j) if (j != im && s[j] > thr) thr = s[j];
    float e[4], den = 0.f;
#pragma unroll
    for (int j = 0; j < 4; ++j) {
        e[j] = (s[j] >= thr) ? expf(s[j] - mx) : 0.0f;
        den += e[j];
    }
    float r = 1.0f / den;
#pragma unroll
    for (int j = 0; j < 4; ++j) w[j] = e[j] * r;
}

// ---------------------------------------------------------------------------
// Kernel 3: routers.  One wave per 16-token tile, 8 waves/block.
// ---------------------------------------------------------------------------
__global__ void __launch_bounds__(256) k_router(
    const float* __restrict__ feat, const __bf16* __restrict__ hbf,
    const __bf16* __restrict__ p_sf,  const float* __restrict__ sf_b,
    const __bf16* __restrict__ p_bp,  const float* __restrict__ bp_b,
    const __bf16* __restrict__ p_br1, const float* __restrict__ br_b1,
    const __bf16* __restrict__ p_br2, const float* __restrict__ br_b2,
    const __bf16* __restrict__ p_ir1, const float* __restrict__ ir_b1,
    const __bf16* __restrict__ p_ir2, const float* __restrict__ ir_b2,
    float* __restrict__ out_gw, float* __restrict__ out_gl,
    float* __restrict__ out_bw, float* __restrict__ out_bl) {
    __shared__ __bf16 sbuf[8][16 * 64];   // stage_emb / bemb (bf16, row-major)
    __shared__ __bf16 hbuf[8][16 * 64];   // hidden MLP activations
    __shared__ float  lbuf[8][16 * 16];   // logits tile staging
    __shared__ float  bwls[8][16 * 4];
    __shared__ float  blls[8][16 * 4];

    int wave = threadIdx.x >> 5, lane = threadIdx.x & 31;
    int tile = blockIdx.x * 8 + wave;
    int t0 = tile * 16;
    int nlo = lane & 15, hh = lane >> 4;

    // ---- stage_emb = feat @ sf_W + sf_b (K=32, N=64) ----
    {
        v16bf a = loadA_f32pad(feat + (size_t)t0 * FTOT, FTOT, lane, 32);
#pragma unroll
        for (int nt = 0; nt < 4; ++nt) {
            v8f c = {};
            c = wmma_bf16(a, loadB(p_sf + nt * 512, lane), c);
            int n = nt * 16 + nlo;
            float bias = sf_b[n];
#pragma unroll
            for (int v = 0; v < 8; ++v)
                sbuf[wave][(hh * 8 + v) * 64 + n] = (__bf16)(c[v] + bias);
        }
    }
    __syncthreads();

    // ---- bundle hidden: gelu([h | stage_emb] @ br_W1 + b1), K=320 ----
#pragma unroll
    for (int nt = 0; nt < 4; ++nt) {
        v8f c = {};
#pragma unroll
        for (int kb = 0; kb < 10; ++kb) {
            v16bf a = (kb < 8) ? loadA_bf16(hbf + (size_t)t0 * DDIM, DDIM, lane, kb)
                               : loadA_bf16(sbuf[wave], 64, lane, kb - 8);
            c = wmma_bf16(a, loadB(p_br1 + (nt * 10 + kb) * 512, lane), c);
        }
        int n = nt * 16 + nlo;
        float bias = br_b1[n];
#pragma unroll
        for (int v = 0; v < 8; ++v)
            hbuf[wave][(hh * 8 + v) * 64 + n] = (__bf16)gelu_erf(c[v] + bias);
    }
    __syncthreads();

    // ---- bundle logits (N padded 4->16) ----
    {
        v8f c = {};
#pragma unroll
        for (int kb = 0; kb < 2; ++kb)
            c = wmma_bf16(loadA_bf16(hbuf[wave], 64, lane, kb),
                          loadB(p_br2 + kb * 512, lane), c);
#pragma unroll
        for (int v = 0; v < 8; ++v)
            lbuf[wave][(hh * 8 + v) * 16 + nlo] = c[v] + (nlo < 4 ? br_b2[nlo] : 0.0f);
    }
    __syncthreads();
    if (lane < 16) {
        float s[4], w[4];
#pragma unroll
        for (int j = 0; j < 4; ++j) s[j] = lbuf[wave][lane * 16 + j];
        top2sm4(s, w);
        int t = t0 + lane;
#pragma unroll
        for (int j = 0; j < 4; ++j) {
            out_bl[(size_t)t * 4 + j] = s[j];
            out_bw[(size_t)t * 4 + j] = w[j];
            blls[wave][lane * 4 + j] = s[j];
            bwls[wave][lane * 4 + j] = w[j];
        }
    }
    __syncthreads();

    // ---- intra-bundle routers ----
    for (int g = 0; g < 4; ++g) {
        // bemb = bfeat_g @ bp_W[g] + bp_b[g]  (K=8 padded to 32)
        v16bf af = loadA_f32pad(feat + (size_t)t0 * FTOT + g * 8, FTOT, lane, 8);
#pragma unroll
        for (int nt = 0; nt < 4; ++nt) {
            v8f c = {};
            c = wmma_bf16(af, loadB(p_bp + (g * 4 + nt) * 512, lane), c);
            int n = nt * 16 + nlo;
            float bias = bp_b[g * 64 + n];
#pragma unroll
            for (int v = 0; v < 8; ++v)
                sbuf[wave][(hh * 8 + v) * 64 + n] = (__bf16)(c[v] + bias);
        }
        __syncthreads();
        // ih = gelu([h | bemb] @ ir_W1[g] + ir_b1[g])
#pragma unroll
        for (int nt = 0; nt < 4; ++nt) {
            v8f c = {};
#pragma unroll
            for (int kb = 0; kb < 10; ++kb) {
                v16bf a = (kb < 8) ? loadA_bf16(hbf + (size_t)t0 * DDIM, DDIM, lane, kb)
                                   : loadA_bf16(sbuf[wave], 64, lane, kb - 8);
                c = wmma_bf16(a, loadB(p_ir1 + ((g * 4 + nt) * 10 + kb) * 512, lane), c);
            }
            int n = nt * 16 + nlo;
            float bias = ir_b1[g * 64 + n];
#pragma unroll
            for (int v = 0; v < 8; ++v)
                hbuf[wave][(hh * 8 + v) * 64 + n] = (__bf16)gelu_erf(c[v] + bias);
        }
        __syncthreads();
        // il = ih @ ir_W2[g] + ir_b2[g]
        {
            v8f c = {};
#pragma unroll
            for (int kb = 0; kb < 2; ++kb)
                c = wmma_bf16(loadA_bf16(hbuf[wave], 64, lane, kb),
                              loadB(p_ir2 + (g * 2 + kb) * 512, lane), c);
#pragma unroll
            for (int v = 0; v < 8; ++v)
                lbuf[wave][(hh * 8 + v) * 16 + nlo] =
                    c[v] + (nlo < 4 ? ir_b2[g * 4 + nlo] : 0.0f);
        }
        __syncthreads();
        if (lane < 16) {
            float s[4], w[4];
#pragma unroll
            for (int j = 0; j < 4; ++j) s[j] = lbuf[wave][lane * 16 + j];
            top2sm4(s, w);
            int t = t0 + lane;
            float bwv = bwls[wave][lane * 4 + g];
            float blv = blls[wave][lane * 4 + g];
#pragma unroll
            for (int j = 0; j < 4; ++j) {
                int e = g * 4 + j;
                out_gl[(size_t)t * NE + e] = blv + s[j];
                out_gw[(size_t)t * NE + e] = bwv * w[j];
            }
        }
        __syncthreads();
    }
}

// ---------------------------------------------------------------------------
// Kernel 4: experts. 8 waves x 16 tokens = 128 tokens/block; 16 experts looped,
// W1/W2 double-buffered in LDS via TDM.
// ---------------------------------------------------------------------------
__global__ void __launch_bounds__(256) k_expert(
    const __bf16* __restrict__ hbf,
    const __bf16* __restrict__ pW1, const __bf16* __restrict__ pW2,
    const float* __restrict__ e_b1, const float* __restrict__ e_b2,
    const float* __restrict__ gwsrc, const float* __restrict__ hidden,
    const float* __restrict__ alpha_p,
    float* __restrict__ out_next, float* __restrict__ out_delta) {
    extern __shared__ char smem[];
    __bf16* w1b = (__bf16*)smem;           // [2][32768]
    __bf16* w2b = w1b + 2 * 32768;         // [2][32768]
    __bf16* h1  = w2b + 2 * 32768;         // [8][16*128]
    float*  gws = (float*)(h1 + 8 * 2048); // [128][16]

    int wave = threadIdx.x >> 5, lane = threadIdx.x & 31;
    int tb = blockIdx.x * 128;
    int nlo = lane & 15, hh = lane >> 4;

    for (int i = threadIdx.x; i < 128 * NE; i += 256)
        gws[i] = gwsrc[(size_t)tb * NE + i];
    if (wave == 0) {                      // prologue prefetch, expert 0
        tdm_copy_64k(pW1, w1b);
        tdm_copy_64k(pW2, w2b);
    }
    __syncthreads();

    int t0 = tb + wave * 16;
    v16bf a1[8];
#pragma unroll
    for (int kb = 0; kb < 8; ++kb)
        a1[kb] = loadA_bf16(hbf + (size_t)t0 * DDIM, DDIM, lane, kb);

    v8f dacc[16];
#pragma unroll
    for (int nt = 0; nt < 16; ++nt) dacc[nt] = (v8f){};

    __bf16* myH = h1 + wave * 2048;

    for (int e = 0; e < NE; ++e) {
        int buf = e & 1;
        if (wave == 0) __builtin_amdgcn_s_wait_tensorcnt(0);
        __syncthreads();   // weights for expert e visible to all waves
        if (wave == 0 && e < NE - 1) {     // overlap next expert's DMA
            tdm_copy_64k(pW1 + (size_t)(e + 1) * 32768, w1b + ((e + 1) & 1) * 32768);
            tdm_copy_64k(pW2 + (size_t)(e + 1) * 32768, w2b + ((e + 1) & 1) * 32768);
        }
        const __bf16* W1 = w1b + buf * 32768;
        const __bf16* W2 = w2b + buf * 32768;

        // GEMM1: [16,256] x [256,128] -> gelu -> LDS (bf16 row-major)
#pragma unroll
        for (int nt = 0; nt < 8; ++nt) {
            v8f c = {};
#pragma unroll
            for (int kb = 0; kb < 8; ++kb)
                c = wmma_bf16(a1[kb], loadB(W1 + (nt * 8 + kb) * 512, lane), c);
            int n = nt * 16 + nlo;
            float b1v = e_b1[e * DEH + n];
#pragma unroll
            for (int v = 0; v < 8; ++v)
                myH[(hh * 8 + v) * DEH + n] = (__bf16)gelu_erf(c[v] + b1v);
        }

        v16bf a2[4];
#pragma unroll
        for (int kb = 0; kb < 4; ++kb)
            a2[kb] = loadA_bf16(myH, DEH, lane, kb);

        float gv[8];
#pragma unroll
        for (int v = 0; v < 8; ++v)
            gv[v] = gws[(wave * 16 + hh * 8 + v) * NE + e];

        // GEMM2: [16,128] x [128,256]; gate-scaled accumulate into dacc
#pragma unroll
        for (int nt = 0; nt < 16; ++nt) {
            v8f c = {};
#pragma unroll
            for (int kb = 0; kb < 4; ++kb)
                c = wmma_bf16(a2[kb], loadB(W2 + (nt * 4 + kb) * 512, lane), c);
            int n = nt * 16 + nlo;
            float b2v = e_b2[e * DDIM + n];
#pragma unroll
            for (int v = 0; v < 8; ++v)
                dacc[nt][v] += gv[v] * (c[v] + b2v);
        }
        __syncthreads();   // all waves done with buf before it is refilled
    }

    float al = alpha_p[0];
#pragma unroll
    for (int nt = 0; nt < 16; ++nt) {
        int n = nt * 16 + nlo;
#pragma unroll
        for (int v = 0; v < 8; ++v) {
            size_t idx = (size_t)(t0 + hh * 8 + v) * DDIM + n;
            float d = dacc[nt][v];
            out_delta[idx] = d;
            out_next[idx]  = hidden[idx] + al * d;
        }
    }
}

// ---------------------------------------------------------------------------
// Host launcher
// ---------------------------------------------------------------------------
extern "C" void kernel_launch(void* const* d_in, const int* in_sizes, int n_in,
                              void* d_out, int out_size, void* d_ws, size_t ws_size,
                              hipStream_t stream) {
    (void)in_sizes; (void)n_in; (void)out_size; (void)ws_size;
    const float* hidden = (const float*)d_in[0];
    const float* feat   = (const float*)d_in[1];
    const float* ln_g   = (const float*)d_in[2];
    const float* ln_b   = (const float*)d_in[3];
    const float* sf_W   = (const float*)d_in[4];
    const float* sf_b   = (const float*)d_in[5];
    const float* bp_W   = (const float*)d_in[6];
    const float* bp_b   = (const float*)d_in[7];
    const float* br_W1  = (const float*)d_in[8];
    const float* br_b1  = (const float*)d_in[9];
    const float* br_W2  = (const float*)d_in[10];
    const float* br_b2  = (const float*)d_in[11];
    const float* ir_W1  = (const float*)d_in[12];
    const float* ir_b1  = (const float*)d_in[13];
    const float* ir_W2  = (const float*)d_in[14];
    const float* ir_b2  = (const float*)d_in[15];
    const float* e_W1   = (const float*)d_in[16];
    const float* e_b1   = (const float*)d_in[17];
    const float* e_W2   = (const float*)d_in[18];
    const float* e_b2   = (const float*)d_in[19];
    const float* alpha  = (const float*)d_in[20];

    float* out = (float*)d_out;
    float* out_next  = out;                 // [16384,256]
    float* out_gw    = out + 4194304;       // [16384,16]
    float* out_gl    = out + 4456448;       // [16384,16]
    float* out_bw    = out + 4718592;       // [16384,4]
    float* out_bl    = out + 4784128;       // [16384,4]
    float* out_delta = out + 4849664;       // [16384,256]

    // workspace carve-up (bf16 regions, all 256B aligned)
    char* ws = (char*)d_ws;
    __bf16* hbf   = (__bf16*)ws;  ws += (size_t)NT * DDIM * 2;   // 8 MB
    __bf16* pW1   = (__bf16*)ws;  ws += 16 * 8 * 8 * 512 * 2;    // 1 MB
    __bf16* pW2   = (__bf16*)ws;  ws += 16 * 16 * 4 * 512 * 2;   // 1 MB
    __bf16* p_sf  = (__bf16*)ws;  ws += 4 * 512 * 2;
    __bf16* p_bp  = (__bf16*)ws;  ws += 16 * 512 * 2;
    __bf16* p_br1 = (__bf16*)ws;  ws += 40 * 512 * 2;
    __bf16* p_ir1 = (__bf16*)ws;  ws += 160 * 512 * 2;
    __bf16* p_br2 = (__bf16*)ws;  ws += 2 * 512 * 2;
    __bf16* p_ir2 = (__bf16*)ws;  ws += 8 * 512 * 2;

    k_ln<<<2048, 256, 0, stream>>>(hidden, ln_g, ln_b, hbf);

    k_pack<<<512, 256, 0, stream>>>(e_W1,  pW1,   256, 128, 8, 8, 16);
    k_pack<<<512, 256, 0, stream>>>(e_W2,  pW2,   128, 256, 4, 16, 16);
    k_pack<<<8,   256, 0, stream>>>(sf_W,  p_sf,  32,  64,  1, 4, 1);
    k_pack<<<8,   256, 0, stream>>>(bp_W,  p_bp,  8,   64,  1, 4, 4);
    k_pack<<<32,  256, 0, stream>>>(br_W1, p_br1, 320, 64, 10, 4, 1);
    k_pack<<<64,  256, 0, stream>>>(ir_W1, p_ir1, 320, 64, 10, 4, 4);
    k_pack<<<4,   256, 0, stream>>>(br_W2, p_br2, 64,  4,   2, 1, 1);
    k_pack<<<4,   256, 0, stream>>>(ir_W2, p_ir2, 64,  4,   2, 1, 4);

    k_router<<<128, 256, 0, stream>>>(feat, hbf,
                                      p_sf, sf_b, p_bp, bp_b,
                                      p_br1, br_b1, p_br2, br_b2,
                                      p_ir1, ir_b1, p_ir2, ir_b2,
                                      out_gw, out_gl, out_bw, out_bl);

    const int expert_lds = 2 * 65536 + 2 * 65536 / 2 /*w2 half?*/;
    // w1: 2*64KB, w2: 2*64KB, h1: 32KB, gw: 8KB  => 303104 bytes
    const int lds_bytes = 2 * 65536 + 2 * 65536 + 8 * 2048 * 2 + 128 * 16 * 4;
    (void)expert_lds;
    hipFuncSetAttribute(reinterpret_cast<const void*>(k_expert),
                        hipFuncAttributeMaxDynamicSharedMemorySize, lds_bytes);
    k_expert<<<128, 256, lds_bytes, stream>>>(hbf, pW1, pW2, e_b1, e_b2,
                                              out_gw, hidden, alpha,
                                              out_next, out_delta);
}
